// HSNLayer_80977313398932
// MI455X (gfx1250) — matrix operand order repaired and checked by reading, not compiled, not run on hardware
//
#include <hip/hip_runtime.h>
#include <hip/hip_bf16.h>
#include <cstdint>

#define C 128
#define N_EDGES_CONST 200000

typedef __attribute__((ext_vector_type(16))) __bf16 v16bf;
typedef __attribute__((ext_vector_type(8)))  __bf16 v8bf;
typedef __attribute__((ext_vector_type(4)))  __bf16 v4bf;
typedef __attribute__((ext_vector_type(8)))  float  v8f;

__device__ __forceinline__ __bf16 f2bf(float f) {
  uint32_t u = __builtin_bit_cast(uint32_t, f);
  u += 0x7FFFu + ((u >> 16) & 1u);            // round-to-nearest-even
  uint16_t h = (uint16_t)(u >> 16);
  return __builtin_bit_cast(__bf16, h);
}

// Native fp32 global atomic (L2-resolved), device scope for cross-WGP adds.
__device__ __forceinline__ void atomic_add_f32_dev(float* p, float v) {
  asm volatile("global_atomic_add_f32 %0, %1, off scope:SCOPE_DEV"
               :: "v"(p), "v"(v) : "memory");
}

// ---------------------------------------------------------------------------
// Y[M x 128] = X[M x 128] @ W[128 x 128]
// bf16 WMMA (v_wmma_f32_16x16x32_bf16), f32 accumulate.
// Block = 256 threads (8 waves); each wave owns a 16-row stripe; block = 128 rows.
// W is staged TRANSPOSED in LDS so B fragments are contiguous -> ds_load_b128.
// ---------------------------------------------------------------------------
#define WS 136   // lds row stride in bf16 elems (+8 pad; 136*2B = 272B -> bank spread)
#define XS 136

__global__ __launch_bounds__(256) void gemm128_wmma(const float* __restrict__ X,
                                                    const float* __restrict__ W,
                                                    float* __restrict__ Y, int M) {
  __shared__ __bf16 lds_wt[128 * WS];   // W transposed: lds_wt[col*WS + k] = W[k][col]
  __shared__ __bf16 lds_x[128 * XS];    // row-major X tile (bf16)

  const int t       = threadIdx.x;
  const int lane    = t & 31;
  const int wave    = t >> 5;
  const int rowBase = blockIdx.x * 128;

  // ---- Stage W^T: 2 threads per column, 64 k's each; packed b128 LDS stores.
  {
    const int c  = t & 127;
    const int k0 = (t >> 7) * 64;
    for (int kb = k0; kb < k0 + 64; kb += 8) {
      v8bf pk;
      #pragma unroll
      for (int j = 0; j < 8; ++j)
        pk[j] = f2bf(W[(size_t)(kb + j) * C + c]);   // coalesced across lanes (same kb+j)
      *(v8bf*)&lds_wt[c * WS + kb] = pk;             // one ds_store_b128
    }
  }
  // ---- Stage X tile (128 rows): float4 global loads, 4xbf16 packed b64 stores.
  for (int idx = t * 4; idx < 128 * 128; idx += 256 * 4) {
    int r = idx >> 7, col = idx & 127;
    int gr = rowBase + r;
    float4 f = (gr < M) ? *(const float4*)&X[(size_t)gr * C + col]
                        : make_float4(0.f, 0.f, 0.f, 0.f);
    v4bf pk;
    pk[0] = f2bf(f.x); pk[1] = f2bf(f.y); pk[2] = f2bf(f.z); pk[3] = f2bf(f.w);
    *(v4bf*)&lds_x[r * XS + col] = pk;               // one ds_store_b64
  }
  __syncthreads();

  const int half = lane >> 4;        // 0: lanes 0-15, 1: lanes 16-31
  const int l15  = lane & 15;
  const int lr   = wave * 16 + l15;  // A-matrix row owned by this lane

  v8f zero = {};
  v8f acc[8];
  #pragma unroll
  for (int nt = 0; nt < 8; ++nt) acc[nt] = zero;

  #pragma unroll
  for (int kt = 0; kt < 4; ++kt) {
    // A fragment 16x32 bf16: lane holds K=[kbA..kbA+7] and K=[kbA+16..kbA+23]
    const int kbA = kt * 32 + half * 8;
    v16bf a;
    {
      v8bf* ap = (v8bf*)&a;
      ap[0] = *(const v8bf*)&lds_x[lr * XS + kbA];        // ds_load_b128
      ap[1] = *(const v8bf*)&lds_x[lr * XS + kbA + 16];   // ds_load_b128
    }
    // B fragment 32x16 bf16: lane = column, K=[kbB..kbB+7] and K=[kbB+16..kbB+23] from W^T
    const int kbB = kt * 32 + half * 8;
    #pragma unroll
    for (int nt = 0; nt < 8; ++nt) {
      const int colB = nt * 16 + l15;
      v16bf b;
      {
        v8bf* bp = (v8bf*)&b;
        bp[0] = *(const v8bf*)&lds_wt[colB * WS + kbB];       // ds_load_b128
        bp[1] = *(const v8bf*)&lds_wt[colB * WS + kbB + 16];  // ds_load_b128
      }
      acc[nt] = __builtin_amdgcn_wmma_f32_16x16x32_bf16(
          false, a, false, b, (short)0, acc[nt], false, false);
    }
  }

  // D layout: lanes 0-15 -> N=lane, VGPR j -> M=j ; lanes 16-31 -> M=j+8
  const int rowOut0 = rowBase + wave * 16 + half * 8;
  #pragma unroll
  for (int nt = 0; nt < 8; ++nt) {
    const int col = nt * 16 + l15;
    #pragma unroll
    for (int j = 0; j < 8; ++j) {
      int row = rowOut0 + j;
      if (row < M) Y[(size_t)row * C + col] = acc[nt][j];
    }
  }
}

// ---------------------------------------------------------------------------
// COO SpMM scatter: out[rows[e], :] += vals[e] * Y[cols[e], :]
// One wave per nonzero: scalar index loads, coalesced 512B row gather,
// 4 native f32 atomics per lane.
// ---------------------------------------------------------------------------
__global__ __launch_bounds__(256) void spmm_scatter(const int* __restrict__ rows,
                                                    const int* __restrict__ cols,
                                                    const float* __restrict__ vals,
                                                    const float* __restrict__ Ysrc,
                                                    float* __restrict__ out, int nnz) {
  long gid = (long)blockIdx.x * 256 + threadIdx.x;
  int e = __builtin_amdgcn_readfirstlane((int)(gid >> 5));  // wave-uniform -> s_loads
  if (e >= nnz) return;
  int lane = (int)(gid & 31);
  int r = rows[e];
  int c = cols[e];
  float v = vals[e];
  const float4* src = (const float4*)(Ysrc + (size_t)c * C);
  float4 p = src[lane];                       // global_load_b128, coalesced 512B/row
  float* dst = out + (size_t)r * C + lane * 4;
  atomic_add_f32_dev(dst + 0, v * p.x);
  atomic_add_f32_dev(dst + 1, v * p.y);
  atomic_add_f32_dev(dst + 2, v * p.z);
  atomic_add_f32_dev(dst + 3, v * p.w);
}

__global__ __launch_bounds__(256) void zero_f4(float4* __restrict__ p, long n4) {
  long i = (long)blockIdx.x * 256 + threadIdx.x;
  if (i < n4) p[i] = make_float4(0.f, 0.f, 0.f, 0.f);
}

__global__ __launch_bounds__(256) void sigmoid_f4(float4* __restrict__ p, long n4) {
  long i = (long)blockIdx.x * 256 + threadIdx.x;
  if (i < n4) {
    float4 v = p[i];
    v.x = 1.f / (1.f + __expf(-v.x));
    v.y = 1.f / (1.f + __expf(-v.y));
    v.z = 1.f / (1.f + __expf(-v.z));
    v.w = 1.f / (1.f + __expf(-v.w));
    p[i] = v;
  }
}

// ---------------------------------------------------------------------------
extern "C" void kernel_launch(void* const* d_in, const int* in_sizes, int n_in,
                              void* d_out, int out_size, void* d_ws, size_t ws_size,
                              hipStream_t stream) {
  const float* x     = (const float*)d_in[0];
  const float* W1_00 = (const float*)d_in[1];
  const float* W1_01 = (const float*)d_in[2];
  const float* W2_00 = (const float*)d_in[3];
  const float* W2_10 = (const float*)d_in[4];
  const int*   adj_r = (const int*)d_in[5];
  const int*   adj_c = (const int*)d_in[6];
  const float* adj_v = (const float*)d_in[7];
  const int*   inc_r = (const int*)d_in[8];
  const int*   inc_c = (const int*)d_in[9];
  const float* inc_v = (const float*)d_in[10];

  const int NN   = in_sizes[0] / C;   // 100000 nodes
  const int NE   = N_EDGES_CONST;     // 200000 edges
  const int annz = in_sizes[5];       // 3.2M
  const int innz = in_sizes[8];       // 400K

  float* ws    = (float*)d_ws;
  float* y_a   = ws;                            // NN*C (y1, later y3)
  float* y_b   = y_a  + (size_t)NN * C;         // NE*C (y2 uses NN rows, y4 uses NE)
  float* acc_n = y_b  + (size_t)NE * C;         // NN*C node accumulator (level 1)
  float* acc_e = acc_n + (size_t)NN * C;        // NE*C edge accumulator (level 1)
  float* out   = (float*)d_out;                 // NN*C

  const long nN4 = (long)NN * C / 4;
  const long nE4 = (long)NE * C / 4;
  auto blocks4 = [](long n4) { return (int)((n4 + 255) / 256); };
  const int gemmBlkNN = (NN + 127) / 128;
  const int gemmBlkNE = (NE + 127) / 128;
  const int spmmBlkA  = (int)(((long)annz * 32 + 255) / 256);
  const int spmmBlkI  = (int)(((long)innz * 32 + 255) / 256);

  // zero accumulators (ws is poisoned; must re-init every call)
  zero_f4<<<blocks4(nN4), 256, 0, stream>>>((float4*)acc_n, nN4);
  zero_f4<<<blocks4(nE4), 256, 0, stream>>>((float4*)acc_e, nE4);
  zero_f4<<<blocks4(nN4), 256, 0, stream>>>((float4*)out,   nN4);

  // Level 1 dense: y1 = x @ W1_00 ; y2 = x @ W1_01
  gemm128_wmma<<<gemmBlkNN, 256, 0, stream>>>(x, W1_00, y_a, NN);
  gemm128_wmma<<<gemmBlkNN, 256, 0, stream>>>(x, W1_01, y_b, NN);

  // Level 1 sparse: nodes->nodes (A0) and nodes->edges (B1^T: rows=inc_cols)
  spmm_scatter<<<spmmBlkA, 256, 0, stream>>>(adj_r, adj_c, adj_v, y_a, acc_n, annz);
  spmm_scatter<<<spmmBlkI, 256, 0, stream>>>(inc_c, inc_r, inc_v, y_b, acc_e, innz);

  // sigmoid updates -> x_nodes_l1 (acc_n), x_edges_l1 (acc_e)
  sigmoid_f4<<<blocks4(nN4), 256, 0, stream>>>((float4*)acc_n, nN4);
  sigmoid_f4<<<blocks4(nE4), 256, 0, stream>>>((float4*)acc_e, nE4);

  // Level 2 dense: y3 = x_nodes_l1 @ W2_00 ; y4 = x_edges_l1 @ W2_10
  gemm128_wmma<<<gemmBlkNN, 256, 0, stream>>>(acc_n, W2_00, y_a, NN);
  gemm128_wmma<<<gemmBlkNE, 256, 0, stream>>>(acc_e, W2_10, y_b, NE);

  // Level 2 sparse: nodes->nodes (A0) and edges->nodes (B1: rows=inc_rows)
  spmm_scatter<<<spmmBlkA, 256, 0, stream>>>(adj_r, adj_c, adj_v, y_a, out, annz);
  spmm_scatter<<<spmmBlkI, 256, 0, stream>>>(inc_r, inc_c, inc_v, y_b, out, innz);

  // final merge sigmoid
  sigmoid_f4<<<blocks4(nN4), 256, 0, stream>>>((float4*)out, nN4);
}
